// PConvLinearOpt_60370060313235
// MI455X (gfx1250) — matrix operand order repaired and checked by reading, not compiled, not run on hardware
//
#include <hip/hip_runtime.h>
#include <hip/hip_bf16.h>

// ---------------- problem constants (from reference) ----------------
#define N_PTS   100000
#define KNB     16
#define C_IN    64
#define C_ADD   16
#define C_TOT   80          // C_IN + C_ADD
#define C_MID   16
#define IN_F    1280        // C_TOT * C_MID
#define OUT_F   256

// ---------------- tiling ----------------
#define TILE_N   16         // points per workgroup
#define THREADS  512        // 16 waves (wave32)
#define KCHUNK   4          // neighbor-k chunk staged in LDS (== WMMA f32 K)
#define PCONV_PAD 8
#define PCONV_LD (IN_F + PCONV_PAD)   // halves per point row in LDS
#define NCHUNK   (IN_F / 32)          // 40 stage-2 K-chunks
#define PACK_FRAGS (16 * NCHUNK * 32) // fragments in packed-B workspace
#define PACK_BYTES ((size_t)PACK_FRAGS * 16 * sizeof(_Float16))  // 655360

typedef _Float16 v16h __attribute__((ext_vector_type(16)));
typedef _Float16 v8h  __attribute__((ext_vector_type(8)));
typedef float    v8f  __attribute__((ext_vector_type(8)));
typedef float    v4f  __attribute__((ext_vector_type(4)));
typedef float    v2f  __attribute__((ext_vector_type(2)));

// ---- one-shot pack: LW f32 [256][1280] -> f16 B-fragments [colt][chunk][lane][16]
__global__ __launch_bounds__(256)
void pack_lw_f16(const float* __restrict__ LW, _Float16* __restrict__ BP)
{
    const int fid  = blockIdx.x * 256 + threadIdx.x;   // fragment id, 0..20479
    const int lane = fid & 31;
    const int chunk = (fid >> 5) % NCHUNK;
    const int colt  = fid / (NCHUNK * 32);
    const int o = colt * 16 + (lane & 15);
    const int k = chunk * 32 + (lane >> 4) * 16;
    const v4f* src = (const v4f*)(LW + (size_t)o * IN_F + k);
    const v4f s0 = src[0], s1 = src[1], s2 = src[2], s3 = src[3];
    v16h d;
    #pragma unroll
    for (int h = 0; h < 4; ++h) {
        d[h]      = (_Float16)s0[h];
        d[h + 4]  = (_Float16)s1[h];
        d[h + 8]  = (_Float16)s2[h];
        d[h + 12] = (_Float16)s3[h];
    }
    *(v16h*)(BP + ((size_t)fid << 4)) = d;
}

template <bool USE_PACK>
__global__ __launch_bounds__(THREADS)
void pconv_linear_fused(const float* __restrict__ IF,   // [N, C_IN]
                        const int*   __restrict__ NB,   // [N, KNB]
                        const float* __restrict__ WN,   // [N, KNB, C_MID]
                        const float* __restrict__ AF,   // [N, KNB, C_ADD]
                        const float* __restrict__ LW,   // [OUT_F, IN_F]
                        const float* __restrict__ BIAS, // [OUT_F]
                        const _Float16* __restrict__ BP,// packed B (or null)
                        float*       __restrict__ OUT)  // [N, OUT_F]
{
    __shared__ __attribute__((aligned(16))) float    s_feat[TILE_N][KCHUNK][C_TOT];
    __shared__ __attribute__((aligned(16))) _Float16 s_pconv[TILE_N][PCONV_LD];
    __shared__ int s_nb[TILE_N][KNB];

    const int t    = threadIdx.x;
    const int wave = t >> 5;        // 0..15: point id (stage1) / col tile (stage2)
    const int lane = t & 31;
    const int hi   = lane >> 4;     // 0: lanes 0-15, 1: lanes 16-31
    const int l15  = lane & 15;
    const int n0   = blockIdx.x * TILE_N;

    // ---- stage neighbor indices for this point tile ----
    if (t < TILE_N * KNB) {
        const int p = t >> 4, k = t & 15;
        s_nb[p][k] = NB[(size_t)(n0 + p) * KNB + k];
    }

    // ---- preload exactly the wn values this lane feeds into B fragments ----
    // B (4x16 f32) layout: VGPR0 -> K = hi*2, VGPR1 -> K = hi*2+1, column N = l15.
    // Fold the lane-dependent 'hi' into the ADDRESS (not a register-array index!)
    // so the unrolled kc loop uses pure constant-register references.
    const int p_wn = wave;
    float b0[KNB / KCHUNK], b1[KNB / KCHUNK];
    {
        const float* wptr = WN + ((size_t)(n0 + p_wn) * KNB + hi * 2) * C_MID + l15;
        #pragma unroll
        for (int kc = 0; kc < KNB / KCHUNK; ++kc) {
            b0[kc] = wptr[(kc * KCHUNK + 0) * C_MID];
            b1[kc] = wptr[(kc * KCHUNK + 1) * C_MID];
        }
    }

    // stage-1 accumulators: C/D layout: VGPR r -> (M=r(+8 hi), N=l15)
    v8f dacc[5];
    #pragma unroll
    for (int cc = 0; cc < 5; ++cc)
        #pragma unroll
        for (int r = 0; r < 8; ++r) dacc[cc][r] = 0.f;

    // staging ownership: lane covers k-row kk_s, 8-float column block c8
    const int kk_s = lane >> 3;          // 0..3
    const int c8   = (lane & 7) * 8;     // 0..56

    // ---- stage 1: pconv(80x16) = feat^T @ wn via V_WMMA_F32_16X16X4_F32 ----
    #pragma unroll
    for (int kc = 0; kc < KNB / KCHUNK; ++kc) {
        __syncthreads();   // protect s_feat reuse
        {
            const int kg = kc * KCHUNK + kk_s;
            const int nb = s_nb[p_wn][kg];
            const v4f* ifv = (const v4f*)(IF + (size_t)nb * C_IN + c8);
            *(v4f*)&s_feat[p_wn][kk_s][c8]     = ifv[0];
            *(v4f*)&s_feat[p_wn][kk_s][c8 + 4] = ifv[1];
            if ((lane & 7) < 2) {
                const int ca = (lane & 7) * 8;
                const v4f* afv = (const v4f*)(AF + ((size_t)(n0 + p_wn) * KNB + kg) * C_ADD + ca);
                *(v4f*)&s_feat[p_wn][kk_s][C_IN + ca]     = afv[0];
                *(v4f*)&s_feat[p_wn][kk_s][C_IN + ca + 4] = afv[1];
            }
        }
        __syncthreads();
        // A (16x4 f32): lanes 0-15 -> VGPR0=K0,VGPR1=K1; lanes 16-31 -> K2,K3; M=l15
        #pragma unroll
        for (int cc = 0; cc < 5; ++cc) {
            v2f a, b;
            a.x = s_feat[p_wn][hi * 2 + 0][cc * 16 + l15];
            a.y = s_feat[p_wn][hi * 2 + 1][cc * 16 + l15];
            b.x = b0[kc];
            b.y = b1[kc];
            dacc[cc] = __builtin_amdgcn_wmma_f32_16x16x4_f32(
                false, a, false, b, (short)0, dacc[cc], false, false);
        }
    }

    __syncthreads();
    // write pconv to LDS as f16, flattened f = c*C_MID + m (reference reshape order)
    #pragma unroll
    for (int cc = 0; cc < 5; ++cc)
        #pragma unroll
        for (int r = 0; r < 8; ++r) {
            const int c = cc * 16 + r + hi * 8;
            s_pconv[p_wn][c * C_MID + l15] = (_Float16)dacc[cc][r];
        }
    __syncthreads();

    // ---- stage 2: OUT_tile(16x256) = pconv(16x1280) @ LW^T, f16 WMMA / f32 acc ----
    const int colt = wave;             // wave's 16-column tile
    const int o    = colt * 16 + l15;  // global output channel
    const int prow = l15;              // A row (M) covered by this lane

    v8f cacc;
    const float bv = BIAS[o];
    #pragma unroll
    for (int r = 0; r < 8; ++r) cacc[r] = bv;

    const _Float16* bp_pack = USE_PACK
        ? BP + ((size_t)(colt * NCHUNK) * 32 + lane) * 16 : nullptr;

    for (int kk = 0; kk < IN_F; kk += 32) {
        // A fragment (16-bit A 16x32 layout): two aligned 16B LDS loads
        const int ka = kk + hi * 16;
        const v8h alo = *(const v8h*)&s_pconv[prow][ka];
        const v8h ahi = *(const v8h*)&s_pconv[prow][ka + 8];
        const v16h afrag = __builtin_shufflevector(alo, ahi,
            0, 1, 2, 3, 4, 5, 6, 7, 8, 9, 10, 11, 12, 13, 14, 15);

        v16h bfrag;
        if constexpr (USE_PACK) {
            bfrag = *(const v16h*)bp_pack;       // one coalesced 32B load
            bp_pack += 32 * 16;                  // next chunk
        } else {
            const v4f* bp4 = (const v4f*)(LW + (size_t)o * IN_F + kk + hi * 16);
            const v4f b0v = bp4[0], b1v = bp4[1], b2v = bp4[2], b3v = bp4[3];
            #pragma unroll
            for (int h = 0; h < 4; ++h) {
                bfrag[h]      = (_Float16)b0v[h];
                bfrag[h + 4]  = (_Float16)b1v[h];
                bfrag[h + 8]  = (_Float16)b2v[h];
                bfrag[h + 12] = (_Float16)b3v[h];
            }
        }

        cacc = __builtin_amdgcn_wmma_f32_16x16x32_f16(
            false, afrag, false, bfrag, (short)0, cacc, false, false);
    }

    // ---- store D: VGPR r -> point row n0+r (+8 hi), 64B coalesced rows ----
    #pragma unroll
    for (int r = 0; r < 8; ++r) {
        const int row = n0 + r + hi * 8;
        OUT[(size_t)row * OUT_F + o] = cacc[r];
    }
}

extern "C" void kernel_launch(void* const* d_in, const int* in_sizes, int n_in,
                              void* d_out, int out_size, void* d_ws, size_t ws_size,
                              hipStream_t stream) {
    const float* IF   = (const float*)d_in[0];
    const int*   NB   = (const int*)  d_in[1];
    // d_in[2..4] (inverse_*) unused by forward
    const float* WN   = (const float*)d_in[5];
    const float* AF   = (const float*)d_in[6];
    const float* LW   = (const float*)d_in[7];
    const float* BIAS = (const float*)d_in[8];
    float* OUT = (float*)d_out;

    dim3 grid(N_PTS / TILE_N);   // 6250, exact
    if (ws_size >= PACK_BYTES) {
        _Float16* BP = (_Float16*)d_ws;
        pack_lw_f16<<<PACK_FRAGS / 256, 256, 0, stream>>>(LW, BP);
        pconv_linear_fused<true><<<grid, THREADS, 0, stream>>>(
            IF, NB, WN, AF, LW, BIAS, BP, OUT);
    } else {
        pconv_linear_fused<false><<<grid, THREADS, 0, stream>>>(
            IF, NB, WN, AF, LW, BIAS, nullptr, OUT);
    }
}
